// FeedForwardMoe_19834158973482
// MI455X (gfx1250) — compile-verified, matrix operand order checked by reading, not run on hardware
//
#include <hip/hip_runtime.h>
#include <hip/hip_bf16.h>
#include <math.h>

#define B_     16
#define C_     320
#define HW_    4096
#define E_     16
#define T_     256
#define INNER_ 1280
#define EPS_   1e-5f

typedef __attribute__((ext_vector_type(16))) __bf16 v16bf;
typedef __attribute__((ext_vector_type(8)))  __bf16 v8bf;
typedef __attribute__((ext_vector_type(8)))  float  v8f;

union FragA { v16bf v; v8bf h[2]; };

__device__ __forceinline__ unsigned short f2bf(float f) {
  unsigned int u = __float_as_uint(f);
  unsigned int r = (u + 0x7FFFu + ((u >> 16) & 1u)) >> 16;  // RNE
  return (unsigned short)r;
}

// ---------------------------------------------------------------- inv perm
__global__ void invperm_k(const int* __restrict__ perm, int* __restrict__ inv) {
  int i = blockIdx.x * 256 + threadIdx.x;
  if (i < HW_) inv[perm[i]] = i;
}

// ------------------------------------------------- pack W1 into B-fragments
// Fragment (kc = K/32 chunk, n16 = N/16 tile): 32 lanes x 16 bf16 contiguous.
// lane<16: N=n16*16+lane, K = kc*32 + {0..7,16..23}
// lane>=16: N=n16*16+lane-16, K = kc*32 + {8..15,24..31}
__global__ void pack_w1_k(const float* __restrict__ W1, unsigned short* __restrict__ W1p) {
  int idx = blockIdx.x * 256 + threadIdx.x;
  if (idx >= E_ * C_ * INNER_) return;
  int j    = idx & 15;
  int lane = (idx >> 4) & 31;
  int frag = idx >> 9;
  int n16  = frag % (INNER_ / 16);
  int tmp  = frag / (INNER_ / 16);
  int kc   = tmp % (C_ / 32);
  int e    = tmp / (C_ / 32);
  int kk   = ((lane >= 16) ? 8 : 0) + ((j < 8) ? j : j + 8);
  int K    = kc * 32 + kk;
  int N    = n16 * 16 + (lane & 15);
  W1p[idx] = f2bf(W1[(e * C_ + K) * INNER_ + N]);
}

__global__ void pack_w2_k(const float* __restrict__ W2, unsigned short* __restrict__ W2p) {
  int idx = blockIdx.x * 256 + threadIdx.x;
  if (idx >= E_ * C_ * INNER_) return;
  int j    = idx & 15;
  int lane = (idx >> 4) & 31;
  int frag = idx >> 9;
  int n16  = frag % (C_ / 16);
  int tmp  = frag / (C_ / 16);
  int kc   = tmp % (INNER_ / 32);
  int e    = tmp / (INNER_ / 32);
  int kk   = ((lane >= 16) ? 8 : 0) + ((j < 8) ? j : j + 8);
  int K    = kc * 32 + kk;
  int N    = n16 * 16 + (lane & 15);
  W2p[idx] = f2bf(W2[(e * INNER_ + K) * C_ + N]);
}

// ------------------------------------- LayerNorm + permute + bf16 pack
__global__ __launch_bounds__(256) void ln_perm_k(
    const float* __restrict__ x, const float* __restrict__ gamma,
    const float* __restrict__ beta, const int* __restrict__ inv,
    unsigned short* __restrict__ Xp) {
  __shared__ float psum[16][17];
  __shared__ float psq[16][17];
  __shared__ float smu[16], srs[16];
  __shared__ unsigned short ntile[16][C_];  // normalized bf16 tile (10 KB)

  int tid = threadIdx.x;
  int sTile = blockIdx.x & 255;
  int b = blockIdx.x >> 8;
  int s0 = sTile * 16;
  int sidx = tid & 15;   // spatial within tile (coalesced loads)
  int cg = tid >> 4;     // channel group: 20 channels each

  const float* xb = x + ((size_t)b * C_) * HW_;
  float vals[20];
  float sum = 0.f, sq = 0.f;
#pragma unroll
  for (int k = 0; k < 20; ++k) {
    int c = cg * 20 + k;
    float v = xb[(size_t)c * HW_ + s0 + sidx];
    vals[k] = v; sum += v; sq += v * v;
  }
  psum[sidx][cg] = sum;
  psq[sidx][cg] = sq;
  __syncthreads();
  if (tid < 16) {
    float s = 0.f, q = 0.f;
#pragma unroll
    for (int g = 0; g < 16; ++g) { s += psum[tid][g]; q += psq[tid][g]; }
    float mu = s * (1.0f / C_);
    float var = q * (1.0f / C_) - mu * mu;
    smu[tid] = mu;
    srs[tid] = rsqrtf(var + EPS_);
  }
  __syncthreads();
  float mu = smu[sidx], rs = srs[sidx];
#pragma unroll
  for (int k = 0; k < 20; ++k) {
    int c = cg * 20 + k;
    float v = (vals[k] - mu) * rs * gamma[c] + beta[c];
    ntile[sidx][c] = f2bf(v);
  }
  __syncthreads();
  // coalesced row copy-out: row s -> Xp row inv[s] (640 contiguous bytes)
  for (int i = tid; i < 16 * (C_ / 8); i += 256) {
    int row = i / (C_ / 8);
    int off = i % (C_ / 8);
    int pos = inv[s0 + row];
    const uint4* srcp = (const uint4*)&ntile[row][off * 8];
    uint4* dstp = (uint4*)(Xp + ((size_t)(b * HW_ + pos)) * C_ + off * 8);
    *dstp = *srcp;
  }
}

// ------------------------------------------------- fused MoE FFN (WMMA)
// grid: B*E*(T/64) blocks, 256 threads (8 waves).
// Wave w: row strip rs=w&3 (16 rows), col half ch=w>>2 (160 of 320 out cols).
__global__ __launch_bounds__(256) void moe_ffn_k(
    const unsigned short* __restrict__ Xp,
    const unsigned short* __restrict__ W1p,
    const unsigned short* __restrict__ W2p,
    const float* __restrict__ b1,
    const float* __restrict__ b2,
    const int* __restrict__ perm,
    const float* __restrict__ x,
    float* __restrict__ out) {
  __shared__ unsigned short xlds[64 * C_];     // 40 KB: X block, row-major
  __shared__ unsigned short hlds[2][64 * 32];  // 2 x 4 KB: hid chunk ping-pong

  int tid = threadIdx.x;
  int blk = blockIdx.x;
  int bRow = blk & 3;
  int e = (blk >> 2) & 15;
  int b = blk >> 6;
  int t0 = bRow * 64;

  {  // X block is a contiguous 40960-byte slab in Xp
    const uint4* src = (const uint4*)(Xp + ((size_t)(b * HW_ + e * T_ + t0)) * C_);
    uint4* dst = (uint4*)xlds;
#pragma unroll
    for (int i = 0; i < 10; ++i) dst[tid + i * 256] = src[tid + i * 256];
  }
  __syncthreads();

  int w = tid >> 5;
  int L = tid & 31;
  int rs = w & 3;
  int ch = w >> 2;
  int nlane = L & 15;
  int kb = (L >= 16) ? 8 : 0;        // K half select for A/B fragments
  int rowA = rs * 16 + nlane;        // A-fragment row for this lane

  v8f acc[10];
#pragma unroll
  for (int i = 0; i < 10; ++i)
#pragma unroll
    for (int r = 0; r < 8; ++r) acc[i][r] = 0.f;

  const unsigned short* w1base = W1p + (size_t)e * (C_ / 32) * (INNER_ / 16) * 512;
  const unsigned short* w2base = W2p + (size_t)e * (INNER_ / 32) * (C_ / 16) * 512;

  for (int j = 0; j < INNER_ / 32; ++j) {
    unsigned short* hbuf = hlds[j & 1];

    // ---- preload this chunk's 10 W2 B-fragments (latency hidden by phase 1)
    v16bf bw[10];
    const unsigned short* w2j = w2base + (size_t)j * (C_ / 16) * 512;
#pragma unroll
    for (int nt = 0; nt < 10; ++nt) {
      bw[nt] = *(const v16bf*)(w2j + ((size_t)(ch * 10 + nt)) * 512 + L * 16);
    }
    if (j + 1 < INNER_ / 32)
      __builtin_prefetch(w2base + (size_t)(j + 1) * (C_ / 16) * 512 +
                             (size_t)ch * 10 * 512, 0, 1);

    // ---- phase 1: hid tile rows rs*16.., cols (j*2+ch)*16.. over K=C ----
    // two interleaved accumulators to break the WMMA RAW chain
    v8f hacc0, hacc1;
#pragma unroll
    for (int r = 0; r < 8; ++r) { hacc0[r] = 0.f; hacc1[r] = 0.f; }
    int n16h = j * 2 + ch;
#pragma unroll
    for (int kc = 0; kc < C_ / 32; kc += 2) {
      FragA a0, a1;
      const v8bf* pa0 = (const v8bf*)((const char*)xlds + rowA * (C_ * 2) + kc * 64 + kb * 2);
      const v8bf* pa1 = (const v8bf*)((const char*)xlds + rowA * (C_ * 2) + (kc + 1) * 64 + kb * 2);
      a0.h[0] = pa0[0]; a0.h[1] = pa0[2];
      a1.h[0] = pa1[0]; a1.h[1] = pa1[2];
      const v16bf* pb0 =
          (const v16bf*)(w1base + ((size_t)(kc * (INNER_ / 16) + n16h)) * 512 + L * 16);
      const v16bf* pb1 =
          (const v16bf*)(w1base + ((size_t)((kc + 1) * (INNER_ / 16) + n16h)) * 512 + L * 16);
      hacc0 = __builtin_amdgcn_wmma_f32_16x16x32_bf16(false, a0.v, false, *pb0,
                                                      (short)0, hacc0, false, false);
      hacc1 = __builtin_amdgcn_wmma_f32_16x16x32_bf16(false, a1.v, false, *pb1,
                                                      (short)0, hacc1, false, false);
    }
    // bias + exact GELU + bf16, store to hid LDS (C-layout: N fixed per lane)
    float bias1 = b1[e * INNER_ + n16h * 16 + nlane];
    int mbase = rs * 16 + kb;
    int colh = ch * 16 + nlane;
#pragma unroll
    for (int r = 0; r < 8; ++r) {
      float v = hacc0[r] + hacc1[r] + bias1;
      v = 0.5f * v * (1.0f + erff(v * 0.70710678118654752f));
      hbuf[(mbase + r) * 32 + colh] = f2bf(v);
    }
    __syncthreads();  // only barrier per chunk (ping-pong covers WAR)

    // ---- phase 2: out tile accumulation over this K=32 chunk ----
    FragA a2;
    {
      const v8bf* pa = (const v8bf*)((const char*)hbuf + rowA * 64 + kb * 2);
      a2.h[0] = pa[0];
      a2.h[1] = pa[2];
    }
#pragma unroll
    for (int nt = 0; nt < 10; ++nt) {
      acc[nt] = __builtin_amdgcn_wmma_f32_16x16x32_bf16(false, a2.v, false, bw[nt],
                                                        (short)0, acc[nt], false, false);
    }
  }

  // ---- epilogue: + b2, un-permute, + residual ----
  int srow[8];
#pragma unroll
  for (int r = 0; r < 8; ++r) {
    int trow = t0 + rs * 16 + kb + r;
    srow[r] = perm[e * T_ + trow];
  }
#pragma unroll
  for (int nt = 0; nt < 10; ++nt) {
    int col = ch * 160 + nt * 16 + nlane;
    float bias2 = b2[e * C_ + col];
    const float* xcol = x + ((size_t)(b * C_ + col)) * HW_;
    float* ocol = out + ((size_t)(b * C_ + col)) * HW_;
#pragma unroll
    for (int r = 0; r < 8; ++r) {
      ocol[srow[r]] = acc[nt][r] + bias2 + xcol[srow[r]];
    }
  }
}

// ---------------------------------------------------------------- launch
static constexpr size_t INV_OFF = 0;
static constexpr size_t XP_OFF  = 16384;
static constexpr size_t W1P_OFF = XP_OFF + (size_t)B_ * HW_ * C_ * 2;
static constexpr size_t W2P_OFF = W1P_OFF + (size_t)E_ * C_ * INNER_ * 2;

extern "C" void kernel_launch(void* const* d_in, const int* in_sizes, int n_in,
                              void* d_out, int out_size, void* d_ws, size_t ws_size,
                              hipStream_t stream) {
  const float* x     = (const float*)d_in[0];
  const float* gamma = (const float*)d_in[1];
  const float* beta  = (const float*)d_in[2];
  const float* W1    = (const float*)d_in[3];
  const float* b1    = (const float*)d_in[4];
  const float* W2    = (const float*)d_in[5];
  const float* b2    = (const float*)d_in[6];
  const int*   perm  = (const int*)d_in[7];
  float* out = (float*)d_out;
  char* ws = (char*)d_ws;

  int* inv            = (int*)(ws + INV_OFF);
  unsigned short* Xp  = (unsigned short*)(ws + XP_OFF);
  unsigned short* W1p = (unsigned short*)(ws + W1P_OFF);
  unsigned short* W2p = (unsigned short*)(ws + W2P_OFF);

  invperm_k<<<HW_ / 256, 256, 0, stream>>>(perm, inv);
  int npack = E_ * C_ * INNER_;
  pack_w1_k<<<(npack + 255) / 256, 256, 0, stream>>>(W1, W1p);
  pack_w2_k<<<(npack + 255) / 256, 256, 0, stream>>>(W2, W2p);
  ln_perm_k<<<B_ * (HW_ / 16), 256, 0, stream>>>(x, gamma, beta, inv, Xp);
  moe_ffn_k<<<B_ * E_ * (T_ / 64), 256, 0, stream>>>(Xp, W1p, W2p, b1, b2, perm, x, out);
}
